// VAELayer_71871982731653
// MI455X (gfx1250) — matrix operand order repaired and checked by reading, not compiled, not run on hardware
//
#include <hip/hip_runtime.h>
#include <hip/hip_bf16.h>
#include <stdint.h>

// ============================================================================
// VAE transformer layer forward for MI455X (gfx1250), wave32 + WMMA bf16.
//
// Roofline: ~300 GFLOP of GEMM/attention vs ~1.5GB HBM traffic (65us @
// 23.3TB/s) -> matrix-core bound. All large matmuls use
// v_wmma_f32_16x16x32_bf16 (bf16 in, f32 accumulate). GEMM waves compute
// 32x64 tiles with an explicitly double-buffered K loop (two fragment
// register sets, two 32-wide K-steps per iteration): next-step fragments are
// in flight while current WMMAs execute and no rotate-copies are needed.
// Attention is flash-style (online softmax) computed as S^T = K*Q^T so
// per-query softmax stats live in lanes and need only shfl_xor(16); P^T
// fragments for the O^T = V^T * P^T WMMA are assembled with cross-lane
// shuffles (no LDS).
// ============================================================================

constexpr int kB      = 16;
constexpr int kN      = 1024;
constexpr int kDIM    = 768;
constexpr int kHEADS  = 12;
constexpr int kENC    = 6;
constexpr int kHD     = 64;
constexpr int kNCODES = 16;
constexpr float kEPS    = 1e-5f;
constexpr float kKLTHR  = 0.125f;
constexpr float kSCALE  = 0.125f;          // HD^-0.5
constexpr int kRows = kB * kN;             // 16384
constexpr int kFF   = 4 * kDIM;            // 3072

typedef __attribute__((ext_vector_type(16))) __bf16 bf16x16;
typedef __attribute__((ext_vector_type(8)))  float  f32x8;

#define WMMA_BF16(Af, Bf, Cacc) \
  __builtin_amdgcn_wmma_f32_16x16x32_bf16(false, (Af), false, (Bf), (short)0, (Cacc), false, false)

// ---------------------------------------------------------------------------
// helpers
// ---------------------------------------------------------------------------
__device__ __forceinline__ __bf16 f2bf(float f) {
  union { float f; uint32_t u; } in; in.f = f;
  uint32_t u = in.u;
  uint32_t r = (u + 0x7FFFu + ((u >> 16) & 1u)) >> 16;   // round-to-nearest-even
  union { uint16_t s; __bf16 b; } out; out.s = (uint16_t)r;
  return out.b;
}

__device__ __forceinline__ uint4 ld16(const __bf16* p) {
  return *reinterpret_cast<const uint4*>(p);
}

// A-matrix fragment (bf16 16x32): per lane two 16B pieces at K offsets k0/k1.
__device__ __forceinline__ bf16x16 load_afrag(const __bf16* rowptr, int k0, int k1) {
  union { bf16x16 v; uint4 u[2]; } f;
  f.u[0] = ld16(rowptr + k0);
  f.u[1] = ld16(rowptr + k1);
  return f.v;
}

// B-matrix fragment (bf16 32x16): per lane one contiguous 32B run.
__device__ __forceinline__ bf16x16 load_bfrag(const __bf16* p) {
  union { bf16x16 v; uint4 u[2]; } f;
  f.u[0] = ld16(p);
  f.u[1] = ld16(p + 8);
  return f.v;
}

__device__ __forceinline__ float log_sigmoid(float x) {
  float t = -fabsf(x);
  float l = log1pf(__expf(t));
  return (x >= 0.f) ? -l : (x - l);
}

// Deterministic hash RNG standing in for jax.random.uniform(key 42).
// (Exact threefry reproduction is not validated in this compile-only loop.)
__device__ __forceinline__ float rng_uniform(uint32_t a, uint32_t b) {
  uint32_t s = a * 0x9E3779B9u + b * 0x85EBCA6Bu + 42u;
  s ^= s >> 16; s *= 0x7FEB352Du;
  s ^= s >> 15; s *= 0x846CA68Bu;
  s ^= s >> 16;
  return (float)(s >> 8) * (1.0f / 16777216.0f);
}

// ---------------------------------------------------------------------------
// elementwise kernels
// ---------------------------------------------------------------------------
__global__ void cvt_f32_to_bf16(const float* __restrict__ in,
                                __bf16* __restrict__ out, long long n) {
  long long i = (long long)blockIdx.x * blockDim.x + threadIdx.x;
  long long stride = (long long)gridDim.x * blockDim.x;
  for (; i < n; i += stride) out[i] = f2bf(in[i]);
}

__global__ void relu_sq_to_bf16(const float* __restrict__ in,
                                __bf16* __restrict__ out, long long n) {
  long long i = (long long)blockIdx.x * blockDim.x + threadIdx.x;
  long long stride = (long long)gridDim.x * blockDim.x;
  for (; i < n; i += stride) {
    float v = fmaxf(in[i], 0.f);
    out[i] = f2bf(v * v);
  }
}

__global__ void add_f32(const float* __restrict__ a, const float* __restrict__ b,
                        float* __restrict__ o, long long n) {
  long long i = (long long)blockIdx.x * blockDim.x + threadIdx.x;
  long long stride = (long long)gridDim.x * blockDim.x;
  for (; i < n; i += stride) o[i] = a[i] + b[i];
}

// rms_norm over `width`, one block per row, write bf16
__global__ __launch_bounds__(256) void rmsnorm_bf16(const float* __restrict__ X,
                                                    __bf16* __restrict__ Y, int width) {
  __shared__ float sred[256];
  const int row = blockIdx.x, tid = threadIdx.x;
  const float* x = X + (size_t)row * width;
  float ss = 0.f;
  for (int c = tid; c < width; c += 256) { float v = x[c]; ss += v * v; }
  sred[tid] = ss; __syncthreads();
  for (int s = 128; s > 0; s >>= 1) { if (tid < s) sred[tid] += sred[tid + s]; __syncthreads(); }
  const float scale = rsqrtf(sred[0] / (float)width + kEPS);
  __bf16* y = Y + (size_t)row * width;
  for (int c = tid; c < width; c += 256) y[c] = f2bf(x[c] * scale);
}

// ---------------------------------------------------------------------------
// RoPE + per-head rms_norm for q,k; transpose v into [B,H,HD,N].
// One 32-lane unit per (b,n,h); lane handles the rope pair (2*lane, 2*lane+1).
// ---------------------------------------------------------------------------
__global__ __launch_bounds__(128) void rope_qkv(
    const float* __restrict__ qr, const float* __restrict__ kr,
    const float* __restrict__ vr,
    const float* __restrict__ cosb, const float* __restrict__ sinb,
    __bf16* __restrict__ Qo, __bf16* __restrict__ Ko, __bf16* __restrict__ Vt) {
  const int unit = blockIdx.x * 4 + (threadIdx.x >> 5);
  const int lane = threadIdx.x & 31;
  if (unit >= kB * kN * kHEADS) return;
  const int h = unit % kHEADS;
  const int n = (unit / kHEADS) % kN;
  const int b = unit / (kHEADS * kN);
  const size_t inoff = ((size_t)b * kN + n) * kDIM + h * kHD;
  const float c = cosb[n * (kHD / 2) + lane];
  const float s = sinb[n * (kHD / 2) + lane];
  const size_t headrow = (((size_t)b * kHEADS + h) * kN + n) * kHD + 2 * lane;

  // q
  {
    float re = qr[inoff + 2 * lane], im = qr[inoff + 2 * lane + 1];
    float a = re * c - im * s, bb = re * s + im * c;
    float ss = a * a + bb * bb;
    #pragma unroll
    for (int o = 16; o > 0; o >>= 1) ss += __shfl_xor(ss, o);
    float sc = rsqrtf(ss * (1.0f / kHD) + kEPS);
    Qo[headrow]     = f2bf(a * sc);
    Qo[headrow + 1] = f2bf(bb * sc);
  }
  // k
  {
    float re = kr[inoff + 2 * lane], im = kr[inoff + 2 * lane + 1];
    float a = re * c - im * s, bb = re * s + im * c;
    float ss = a * a + bb * bb;
    #pragma unroll
    for (int o = 16; o > 0; o >>= 1) ss += __shfl_xor(ss, o);
    float sc = rsqrtf(ss * (1.0f / kHD) + kEPS);
    Ko[headrow]     = f2bf(a * sc);
    Ko[headrow + 1] = f2bf(bb * sc);
  }
  // v transpose -> Vt[b][h][d][n]
  {
    const size_t vtb = ((size_t)b * kHEADS + h) * kHD * kN;
    Vt[vtb + (size_t)(2 * lane) * kN + n]     = f2bf(vr[inoff + 2 * lane]);
    Vt[vtb + (size_t)(2 * lane + 1) * kN + n] = f2bf(vr[inoff + 2 * lane + 1]);
  }
}

// ---------------------------------------------------------------------------
// Generic NT GEMM: C[M,N] f32 = A[M,K]bf16 * W[N,K]bf16^T.   Requires K%64==0.
// 1 wave -> 32x64 tile; explicit double buffering: two fragment register
// sets, two 32-wide K-steps per iteration, no rotate-copies.
// ---------------------------------------------------------------------------
__global__ __launch_bounds__(256) void gemm_nt_wmma(
    const __bf16* __restrict__ A, int lda,
    const __bf16* __restrict__ W, int ldw,
    float* __restrict__ C, int ldc,
    int M, int N, int K) {
  const int lane = threadIdx.x & 31;
  const int lq = lane & 15;
  const int hi = lane >> 4;
  const int tilesN = N >> 6;
  const int tiles = (M >> 5) * tilesN;
  const int wid = blockIdx.x * 8 + (threadIdx.x >> 5);
  if (wid >= tiles) return;
  const int mbase = (wid / tilesN) << 5;
  const int nbase = (wid % tilesN) << 6;

  f32x8 acc0[4], acc1[4];
  #pragma unroll
  for (int t = 0; t < 4; ++t)
    #pragma unroll
    for (int r = 0; r < 8; ++r) { acc0[t][r] = 0.f; acc1[t][r] = 0.f; }

  const __bf16* arow0 = A + (size_t)(mbase + lq) * lda;
  const __bf16* arow1 = A + (size_t)(mbase + 16 + lq) * lda;
  const __bf16* w0 = W + (size_t)(nbase +  0 + lq) * ldw + hi * 16;
  const __bf16* w1 = W + (size_t)(nbase + 16 + lq) * ldw + hi * 16;
  const __bf16* w2 = W + (size_t)(nbase + 32 + lq) * ldw + hi * 16;
  const __bf16* w3 = W + (size_t)(nbase + 48 + lq) * ldw + hi * 16;
  const int alo = hi * 16, ahi = hi * 16 + 8;

  // ---- buffer set 0: preload k = 0 ----
  bf16x16 pa0 = load_afrag(arow0, alo, ahi);
  bf16x16 pa1 = load_afrag(arow1, alo, ahi);
  bf16x16 pb0 = load_bfrag(w0);
  bf16x16 pb1 = load_bfrag(w1);
  bf16x16 pb2 = load_bfrag(w2);
  bf16x16 pb3 = load_bfrag(w3);

  for (int k0 = 0; k0 < K; k0 += 64) {       // K % 64 == 0
    // ---- buffer set 1 <- k0+32 (always in range) ----
    const int k1 = k0 + 32;
    const bf16x16 qa0 = load_afrag(arow0, k1 + alo, k1 + ahi);
    const bf16x16 qa1 = load_afrag(arow1, k1 + alo, k1 + ahi);
    const bf16x16 qb0 = load_bfrag(w0 + k1);
    const bf16x16 qb1 = load_bfrag(w1 + k1);
    const bf16x16 qb2 = load_bfrag(w2 + k1);
    const bf16x16 qb3 = load_bfrag(w3 + k1);
    __builtin_prefetch(arow0 + k0 + 160, 0, 3);

    // ---- consume set 0 (k0) ----
    acc0[0] = WMMA_BF16(pa0, pb0, acc0[0]);
    acc0[1] = WMMA_BF16(pa0, pb1, acc0[1]);
    acc0[2] = WMMA_BF16(pa0, pb2, acc0[2]);
    acc0[3] = WMMA_BF16(pa0, pb3, acc0[3]);
    acc1[0] = WMMA_BF16(pa1, pb0, acc1[0]);
    acc1[1] = WMMA_BF16(pa1, pb1, acc1[1]);
    acc1[2] = WMMA_BF16(pa1, pb2, acc1[2]);
    acc1[3] = WMMA_BF16(pa1, pb3, acc1[3]);

    // ---- buffer set 0 <- k0+64 (guarded) ----
    const int k2 = k0 + 64;
    if (k2 < K) {
      pa0 = load_afrag(arow0, k2 + alo, k2 + ahi);
      pa1 = load_afrag(arow1, k2 + alo, k2 + ahi);
      pb0 = load_bfrag(w0 + k2);
      pb1 = load_bfrag(w1 + k2);
      pb2 = load_bfrag(w2 + k2);
      pb3 = load_bfrag(w3 + k2);
    }

    // ---- consume set 1 (k0+32) ----
    acc0[0] = WMMA_BF16(qa0, qb0, acc0[0]);
    acc0[1] = WMMA_BF16(qa0, qb1, acc0[1]);
    acc0[2] = WMMA_BF16(qa0, qb2, acc0[2]);
    acc0[3] = WMMA_BF16(qa0, qb3, acc0[3]);
    acc1[0] = WMMA_BF16(qa1, qb0, acc1[0]);
    acc1[1] = WMMA_BF16(qa1, qb1, acc1[1]);
    acc1[2] = WMMA_BF16(qa1, qb2, acc1[2]);
    acc1[3] = WMMA_BF16(qa1, qb3, acc1[3]);
  }

  const int mrow0 = mbase + hi * 8;
  const int mrow1 = mbase + 16 + hi * 8;
  #pragma unroll
  for (int r = 0; r < 8; ++r) {
    float* c0 = C + (size_t)(mrow0 + r) * ldc + nbase + lq;
    c0[0]  = acc0[0][r];
    c0[16] = acc0[1][r];
    c0[32] = acc0[2][r];
    c0[48] = acc0[3][r];
    float* c1 = C + (size_t)(mrow1 + r) * ldc + nbase + lq;
    c1[0]  = acc1[0][r];
    c1[16] = acc1[1][r];
    c1[32] = acc1[2][r];
    c1[48] = acc1[3][r];
  }
}

// ---------------------------------------------------------------------------
// Flash attention: 1 wave per (b,h,16-query tile).
// S^T = K_tile(A) x Q^T(B); online softmax across j; O^T += V^T(A) x P^T(B).
// V^T fragments are loaded at the top of each j-block so they overlap the
// S^T WMMAs and the softmax VALU work.
// enc heads (h < kENC): full attention. dec heads: causal.
// ---------------------------------------------------------------------------
__global__ __launch_bounds__(256) void attn_wmma(
    const __bf16* __restrict__ Q,   // [B,H,N,HD]
    const __bf16* __restrict__ Kh,  // [B,H,N,HD]
    const __bf16* __restrict__ Vt,  // [B,H,HD,N]
    __bf16* __restrict__ Out) {     // [B,N,H*HD]
  const int lane = threadIdx.x & 31;
  const int lq = lane & 15;
  const int hi = lane >> 4;
  const int wid = blockIdx.x * 8 + (threadIdx.x >> 5);
  const int totalWaves = kB * kHEADS * (kN / 16);
  if (wid >= totalWaves) return;
  const int qt = wid & 63;
  const int h  = (wid >> 6) % kHEADS;
  const int b  = wid / (64 * kHEADS);
  const int qbase = qt << 4;
  const int qidx = qbase + lq;
  const bool causal = (h >= kENC);
  const int jend = causal ? (qbase + 16) : kN;

  const size_t headoff = ((size_t)b * kHEADS + h) * kN;
  const __bf16* qp = Q + (headoff + qbase + lq) * kHD;
  const bf16x16 qb0 = load_afrag(qp,      hi * 8, 16 + hi * 8);   // d 0..31
  const bf16x16 qb1 = load_afrag(qp, 32 + hi * 8, 48 + hi * 8);   // d 32..63
  const __bf16* kbase = Kh + headoff * kHD;
  const __bf16* vrow0 = Vt + (((size_t)b * kHEADS + h) * kHD + 0  + lq) * kN;
  const __bf16* vrow1 = Vt + (((size_t)b * kHEADS + h) * kHD + 16 + lq) * kN;
  const __bf16* vrow2 = Vt + (((size_t)b * kHEADS + h) * kHD + 32 + lq) * kN;
  const __bf16* vrow3 = Vt + (((size_t)b * kHEADS + h) * kHD + 48 + lq) * kN;

  f32x8 o[4];
  #pragma unroll
  for (int t = 0; t < 4; ++t)
    #pragma unroll
    for (int r = 0; r < 8; ++r) o[t][r] = 0.f;
  float m = -1e30f, lsum = 0.f;

  for (int j0 = 0; j0 < jend; j0 += 32) {
    // ---- issue all loads for this j-block up front ----
    const __bf16* kr0 = kbase + (size_t)(j0 + lq) * kHD;
    const __bf16* kr1 = kbase + (size_t)(j0 + 16 + lq) * kHD;
    const bf16x16 ka00 = load_afrag(kr0,      hi * 8, 16 + hi * 8);
    const bf16x16 ka01 = load_afrag(kr0, 32 + hi * 8, 48 + hi * 8);
    const bf16x16 ka10 = load_afrag(kr1,      hi * 8, 16 + hi * 8);
    const bf16x16 ka11 = load_afrag(kr1, 32 + hi * 8, 48 + hi * 8);
    const int jlo = j0 + hi * 16, jhi = j0 + hi * 16 + 8;
    const bf16x16 va0 = load_afrag(vrow0, jlo, jhi);   // overlap with S^T wmma
    const bf16x16 va1 = load_afrag(vrow1, jlo, jhi);
    const bf16x16 va2 = load_afrag(vrow2, jlo, jhi);
    const bf16x16 va3 = load_afrag(vrow3, jlo, jhi);

    f32x8 st0, st1;
    #pragma unroll
    for (int r = 0; r < 8; ++r) { st0[r] = 0.f; st1[r] = 0.f; }
    st0 = WMMA_BF16(ka00, qb0, st0);
    st0 = WMMA_BF16(ka01, qb1, st0);
    st1 = WMMA_BF16(ka10, qb0, st1);
    st1 = WMMA_BF16(ka11, qb1, st1);

    // mask + online softmax (each lane owns query row qidx; its 16 j-values
    // cover half of the 32-wide j block; partner half is at lane^16)
    float p0[8], p1[8];
    float bm = -1e30f;
    #pragma unroll
    for (int r = 0; r < 8; ++r) {
      const int ja = j0 + r + hi * 8;
      const int jb = ja + 16;
      float s0 = st0[r] * kSCALE;
      float s1 = st1[r] * kSCALE;
      if (causal && ja > qidx) s0 = -1e30f;
      if (causal && jb > qidx) s1 = -1e30f;
      p0[r] = s0; p1[r] = s1;
      bm = fmaxf(bm, fmaxf(s0, s1));
    }
    bm = fmaxf(bm, __shfl_xor(bm, 16));
    const float mnew = fmaxf(m, bm);
    const float alpha = __expf(m - mnew);
    float ps = 0.f;
    #pragma unroll
    for (int r = 0; r < 8; ++r) {
      p0[r] = __expf(p0[r] - mnew);
      p1[r] = __expf(p1[r] - mnew);
      ps += p0[r] + p1[r];
    }
    lsum = lsum * alpha + ps + __shfl_xor(ps, 16);
    #pragma unroll
    for (int t = 0; t < 4; ++t)
      #pragma unroll
      for (int r = 0; r < 8; ++r) o[t][r] *= alpha;
    m = mnew;

    // build P^T B-fragment (32 j-rows x 16 q-cols) via shfl_xor(16)
    bf16x16 pb;
    #pragma unroll
    for (int r = 0; r < 8; ++r) {
      const float a  = p0[r], bb = p1[r];
      const float ax = __shfl_xor(a, 16);
      const float bx = __shfl_xor(bb, 16);
      pb[r]     = f2bf(hi ? bx : a);   // K rows {0..7} of lane's j-range
      pb[8 + r] = f2bf(hi ? bb : ax);  // K rows {8..15}
    }

    // O^T += V^T x P^T  (4 d-tiles of 16)
    o[0] = WMMA_BF16(va0, pb, o[0]);
    o[1] = WMMA_BF16(va1, pb, o[1]);
    o[2] = WMMA_BF16(va2, pb, o[2]);
    o[3] = WMMA_BF16(va3, pb, o[3]);
  }

  const float inv = 1.0f / lsum;
  __bf16* op = Out + ((size_t)b * kN + qidx) * kDIM + h * kHD + hi * 8;
  #pragma unroll
  for (int r = 0; r < 8; ++r) {
    op[ 0 + r] = f2bf(o[0][r] * inv);
    op[16 + r] = f2bf(o[1][r] * inv);
    op[32 + r] = f2bf(o[2][r] * inv);
    op[48 + r] = f2bf(o[3][r] * inv);
  }
}

// ---------------------------------------------------------------------------
// Latent head: posterior/prior logits (N=4), sampling, KL, z-one-hot,
// xmid = x + dec_out + W_z[:,idx].   One block per (b,n) row.
// ---------------------------------------------------------------------------
__global__ __launch_bounds__(256) void latent_kernel(
    const float* __restrict__ x, const float* __restrict__ enc,
    const float* __restrict__ dec,
    const float* __restrict__ Wpost, const float* __restrict__ bpost,
    const float* __restrict__ Wprior, const float* __restrict__ bprior,
    const float* __restrict__ Wz,
    float* __restrict__ xmid,
    float* __restrict__ out_kl, float* __restrict__ out_klraw,
    float* __restrict__ out_zoh) {
  __shared__ float sred[256];
  __shared__ float res[8];
  __shared__ int sidx;
  const int row = blockIdx.x, tid = threadIdx.x;
  const float* e = enc + (size_t)row * kDIM;
  const float* d = dec + (size_t)row * kDIM;

  float part[8] = {0, 0, 0, 0, 0, 0, 0, 0};
  for (int c = tid; c < kDIM; c += 256) {
    const float ev = e[c], dv = d[c];
    #pragma unroll
    for (int bit = 0; bit < 4; ++bit) {
      part[bit]     += ev * Wpost[bit * kDIM + c];
      part[4 + bit] += dv * Wprior[bit * kDIM + c];
    }
  }
  for (int qd = 0; qd < 8; ++qd) {
    sred[tid] = part[qd]; __syncthreads();
    for (int s = 128; s > 0; s >>= 1) { if (tid < s) sred[tid] += sred[tid + s]; __syncthreads(); }
    if (tid == 0) res[qd] = sred[0];
    __syncthreads();
  }
  if (tid == 0) {
    float klraw = 0.f;
    int idx = 0;
    #pragma unroll
    for (int bit = 0; bit < 4; ++bit) {
      const float post  = res[bit]     + bpost[bit];
      const float prior = res[4 + bit] + bprior[bit];
      const float prob  = 1.f / (1.f + __expf(-post));   // TEMP = 1
      const float u = rng_uniform((uint32_t)row, (uint32_t)bit);
      if (u <= prob) idx |= (1 << bit);
      klraw += prob * (log_sigmoid(post) - log_sigmoid(prior)) +
               (1.f - prob) * (log_sigmoid(-post) - log_sigmoid(-prior));
    }
    out_klraw[row] = klraw;
    out_kl[row] = fmaxf(klraw - kKLTHR, 0.f);
    sidx = idx;
  }
  __syncthreads();
  const int idx = sidx;
  if (tid < kNCODES) out_zoh[(size_t)row * kNCODES + tid] = (tid == idx) ? 1.f : 0.f;
  const float* xr = x + (size_t)row * kDIM;
  float* xo = xmid + (size_t)row * kDIM;
  for (int c = tid; c < kDIM; c += 256)
    xo[c] = xr[c] + d[c] + Wz[(size_t)c * kNCODES + idx];   // z_hidden = W_z[:,idx]
}

// ---------------------------------------------------------------------------
// host orchestration
// ---------------------------------------------------------------------------
extern "C" void kernel_launch(void* const* d_in, const int* in_sizes, int n_in,
                              void* d_out, int out_size, void* d_ws, size_t ws_size,
                              hipStream_t stream) {
  (void)in_sizes; (void)n_in; (void)out_size; (void)ws_size;
  const float* x      = (const float*)d_in[0];
  const float* ropec  = (const float*)d_in[1];
  const float* ropes  = (const float*)d_in[2];
  const float* Wq     = (const float*)d_in[3];
  const float* Wk     = (const float*)d_in[4];
  const float* Wv     = (const float*)d_in[5];
  const float* Woe    = (const float*)d_in[6];
  const float* Wod    = (const float*)d_in[7];
  const float* Wpost  = (const float*)d_in[8];
  const float* bpost  = (const float*)d_in[9];
  const float* Wprior = (const float*)d_in[10];
  const float* bprior = (const float*)d_in[11];
  const float* Wz     = (const float*)d_in[12];
  const float* Wfc1   = (const float*)d_in[13];
  const float* Wfc2   = (const float*)d_in[14];

  // ---- workspace bump allocator (256B aligned) ----
  char* base = (char*)d_ws;
  size_t off = 0;
  auto alloc = [&](size_t bytes) -> void* {
    off = (off + 255) & ~(size_t)255;
    void* p = base + off;
    off += bytes;
    return p;
  };
  const size_t AROW = (size_t)kRows * kDIM;          // 12,582,912 elements
  __bf16* wqb   = (__bf16*)alloc((size_t)kDIM * kDIM * 2);
  __bf16* wkb   = (__bf16*)alloc((size_t)kDIM * kDIM * 2);
  __bf16* wvb   = (__bf16*)alloc((size_t)kDIM * kDIM * 2);
  __bf16* woeb  = (__bf16*)alloc((size_t)kDIM * kDIM * 2);
  __bf16* wodb  = (__bf16*)alloc((size_t)kDIM * (kDIM / 2) * 2);
  __bf16* wfc1b = (__bf16*)alloc((size_t)kFF * kDIM * 2);
  __bf16* wfc2b = (__bf16*)alloc((size_t)kDIM * kFF * 2);
  __bf16* xn    = (__bf16*)alloc(AROW * 2);
  float*  qraw  = (float*)alloc(AROW * 4);
  float*  kraw  = (float*)alloc(AROW * 4);
  float*  vraw  = (float*)alloc(AROW * 4);
  __bf16* qb    = (__bf16*)alloc(AROW * 2 * 4);      // qb|kb|vtb|outb contiguous
  __bf16* kb    = qb + AROW;
  __bf16* vtb   = kb + AROW;
  __bf16* outb  = vtb + AROW;
  float*  act   = (float*)alloc((size_t)kRows * kFF * 4);
  // aliases (lifetime-disjoint reuse)
  float*  enc_out = qraw;          // after rope, qraw is dead
  float*  dec_out = kraw;
  float*  xmid    = vraw;
  __bf16* hbf     = xn;            // xn dead after QKV GEMMs
  __bf16* actb    = qb;            // q/k/vt/out bf16 dead after attention+proj
  float*  fc2out  = kraw;          // dec_out dead after latent kernel

  float* out_x     = (float*)d_out;
  float* out_kl    = out_x + AROW;
  float* out_klraw = out_kl + kRows;
  float* out_zoh   = out_klraw + kRows;

  const int EW = 4096;  // grid-stride elementwise blocks

  // 1) weights -> bf16
  cvt_f32_to_bf16<<<EW, 256, 0, stream>>>(Wq,   wqb,   (long long)kDIM * kDIM);
  cvt_f32_to_bf16<<<EW, 256, 0, stream>>>(Wk,   wkb,   (long long)kDIM * kDIM);
  cvt_f32_to_bf16<<<EW, 256, 0, stream>>>(Wv,   wvb,   (long long)kDIM * kDIM);
  cvt_f32_to_bf16<<<EW, 256, 0, stream>>>(Woe,  woeb,  (long long)kDIM * kDIM);
  cvt_f32_to_bf16<<<EW, 256, 0, stream>>>(Wod,  wodb,  (long long)kDIM * (kDIM / 2));
  cvt_f32_to_bf16<<<EW, 256, 0, stream>>>(Wfc1, wfc1b, (long long)kFF * kDIM);
  cvt_f32_to_bf16<<<EW, 256, 0, stream>>>(Wfc2, wfc2b, (long long)kDIM * kFF);

  // 2) x_norm (bf16)
  rmsnorm_bf16<<<kRows, 256, 0, stream>>>(x, xn, kDIM);

  // 3) QKV projections
  auto gemm = [&](const __bf16* A, int lda, const __bf16* W, int ldw,
                  float* C, int M, int N, int K) {
    const int tiles = (M / 32) * (N / 64);
    gemm_nt_wmma<<<(tiles + 7) / 8, 256, 0, stream>>>(A, lda, W, ldw, C, N, M, N, K);
  };
  gemm(xn, kDIM, wqb, kDIM, qraw, kRows, kDIM, kDIM);
  gemm(xn, kDIM, wkb, kDIM, kraw, kRows, kDIM, kDIM);
  gemm(xn, kDIM, wvb, kDIM, vraw, kRows, kDIM, kDIM);

  // 4) rope + head rmsnorm + v transpose
  rope_qkv<<<(kB * kN * kHEADS) / 4, 128, 0, stream>>>(qraw, kraw, vraw, ropec, ropes,
                                                       qb, kb, vtb);

  // 5) attention (flash, WMMA)
  attn_wmma<<<(kB * kHEADS * (kN / 16)) / 8, 256, 0, stream>>>(qb, kb, vtb, outb);

  // 6/7) output projections (dec uses columns 384..767 = heads 6..11)
  gemm(outb,       kDIM, woeb, kDIM,     enc_out, kRows, kDIM, kDIM);
  gemm(outb + 384, kDIM, wodb, kDIM / 2, dec_out, kRows, kDIM, kDIM / 2);

  // 8) latent: logits, sampling, KL, z-one-hot, xmid
  latent_kernel<<<kRows, 256, 0, stream>>>(x, enc_out, dec_out, Wpost, bpost,
                                           Wprior, bprior, Wz, xmid,
                                           out_kl, out_klraw, out_zoh);

  // 9-13) MLP: h = rmsnorm(xmid); x_out = xmid + relu(h W1^T)^2 W2^T
  rmsnorm_bf16<<<kRows, 256, 0, stream>>>(xmid, hbf, kDIM);
  gemm(hbf, kDIM, wfc1b, kDIM, act, kRows, kFF, kDIM);
  relu_sq_to_bf16<<<EW, 256, 0, stream>>>(act, actb, (long long)kRows * kFF);
  gemm(actb, kFF, wfc2b, kFF, fc2out, kRows, kDIM, kFF);
  add_f32<<<EW, 256, 0, stream>>>(xmid, fc2out, out_x, (long long)kRows * kDIM);
}
